// myBadTransfomerBlock_45191645889033
// MI455X (gfx1250) — matrix operand order repaired and checked by reading, not compiled, not run on hardware
//
#include <hip/hip_runtime.h>

typedef __attribute__((ext_vector_type(16))) __bf16        v16bf;
typedef __attribute__((ext_vector_type(8)))  float         v8f;
typedef __attribute__((ext_vector_type(4)))  unsigned int  v4u;
typedef __attribute__((ext_vector_type(4)))  int           v4i;

// LDS (address_space(3)) qualified types: force ds_* datapath, never flat.
typedef __attribute__((address_space(3))) char lchar;
typedef __attribute__((address_space(3))) v4u  lv4u;
typedef __attribute__((address_space(3))) v4i  lv4i;
typedef __attribute__((address_space(1))) v4i  gv4i;

union Frag {
  v4u   u[2];
  v16bf v;
};

enum { F_RELU = 1, F_OBF16 = 2, F_TRANS = 4 };

#if defined(__has_builtin)
#  if __has_builtin(__builtin_amdgcn_global_load_async_to_lds_b128)
#    define HAVE_ASYNC_LDS 1
#  endif
#  if __has_builtin(__builtin_amdgcn_s_wait_asynccnt)
#    define HAVE_WAIT_ASYNC 1
#  endif
#endif
#ifndef HAVE_ASYNC_LDS
#  define HAVE_ASYNC_LDS 0
#endif
#ifndef HAVE_WAIT_ASYNC
#  define HAVE_WAIT_ASYNC 0
#endif

#if HAVE_ASYNC_LDS
#  if HAVE_WAIT_ASYNC
#    define WAIT_ASYNC(n) __builtin_amdgcn_s_wait_asynccnt(n)
#  else
#    define WAIT_ASYNC(n) asm volatile("s_wait_asynccnt %0" :: "i"(n) : "memory")
#  endif
#else
#  define WAIT_ASYNC(n) ((void)0)
#endif

// 16-byte global -> LDS copy; async (ASYNCcnt-tracked) when the builtin exists.
__device__ __forceinline__ void copy16(const char* g, lchar* l) {
#if HAVE_ASYNC_LDS
  __builtin_amdgcn_global_load_async_to_lds_b128((gv4i*)g, (lv4i*)l, 0, 0);
#else
  *(lv4u*)l = *(const v4u*)g;
#endif
}

__device__ __forceinline__ unsigned short f32_to_bf16(float f) {
  union { float f; unsigned int u; } c; c.f = f;
  unsigned int r = c.u + 0x7FFFu + ((c.u >> 16) & 1u);   // round-to-nearest-even
  return (unsigned short)(r >> 16);
}

// D[M,N] = act( A[M,K] * B[N,K]^T ), A/B bf16 row-major with row length K.
// Block = 8 waves (4 M-groups x 2 N-groups), wave tile 32x64, block tile 128x128.
// K staged through LDS in slices of 64, double-buffered via async-to-LDS.
// M, N must be multiples of 128; K a multiple of 64 (and >= 128).
__global__ __launch_bounds__(256)
void gemm_nt(const unsigned short* __restrict__ A,
             const unsigned short* __restrict__ Bw,
             void* __restrict__ Dv,
             int K,
             long sAb, long sBb, long sDb,   // per-batch strides in elements
             int ldd, int flags)
{
  constexpr int ROWB  = 144;          // 128B of k-data + 16B pad (bank-conflict-free b128)
  constexpr int TILEB = 128 * ROWB;   // 18432 B per tile buffer
  __shared__ v4u smem[(4 * TILEB) / 16];   // A0 A1 B0 B1
  lchar* lbase = (lchar*)smem;             // single AS3 base; all selects are int offsets

  const int bz = blockIdx.z;
  const char* Ag = (const char*)(A  + (long)bz * sAb);
  const char* Bg = (const char*)(Bw + (long)bz * sBb);

  const int tid    = threadIdx.x;
  const int lane   = tid & 31;
  const int wave   = tid >> 5;
  const int lane16 = lane & 15;
  const int hseg   = lane >> 4;              // which half of the wave (0/1)
  const int mW     = (wave & 3) * 32;
  const int nW     = (wave >> 2) * 64;

  const int  tileM0   = blockIdx.y * 128;
  const int  tileN0   = blockIdx.x * 128;
  const long rowBytes = (long)K * 2;

  // Tile copy: 128 rows x 128B = 1024 chunks of 16B per matrix; 4 per thread each.
  int crow[4], ccol[4];
  #pragma unroll
  for (int q = 0; q < 4; ++q) {
    const int c = tid + q * 256;
    crow[q] = c >> 3;            // 8 chunks per row
    ccol[q] = (c & 7) * 16;      // byte offset within the 128B row slice
  }

  auto issue_stage = [&](int s, int bufOff) {
    const long kByte = (long)s * 128;   // 64 k-elements * 2B
    #pragma unroll
    for (int q = 0; q < 4; ++q)
      copy16(Ag + (long)(tileM0 + crow[q]) * rowBytes + kByte + ccol[q],
             lbase + bufOff + crow[q] * ROWB + ccol[q]);
    #pragma unroll
    for (int q = 0; q < 4; ++q)
      copy16(Bg + (long)(tileN0 + crow[q]) * rowBytes + kByte + ccol[q],
             lbase + 2 * TILEB + bufOff + crow[q] * ROWB + ccol[q]);
  };

  v8f acc[2][4];
  #pragma unroll
  for (int i = 0; i < 2; ++i)
    #pragma unroll
    for (int j = 0; j < 4; ++j)
      #pragma unroll
      for (int e = 0; e < 8; ++e)
        acc[i][j][e] = 0.0f;

  auto do_stage = [&](int bufOff) {
    const lchar* Ab3 = lbase + bufOff;
    const lchar* Bb3 = lbase + 2 * TILEB + bufOff;
    #pragma unroll
    for (int j = 0; j < 2; ++j) {        // two k=32 sub-steps per stage
      Frag af[2], bfr[4];
      #pragma unroll
      for (int mi = 0; mi < 2; ++mi) {
        // A 16x32 frag: lanes 0-15 row=lane16 K 0..7 & 16..23; lanes 16-31 K 8..15 & 24..31
        const lchar* p = Ab3 + (mW + mi * 16 + lane16) * ROWB + j * 64 + hseg * 16;
        af[mi].u[0] = *(const lv4u*)(p);
        af[mi].u[1] = *(const lv4u*)(p + 32);
      }
      #pragma unroll
      for (int ni = 0; ni < 4; ++ni) {
        // B 32x16 frag: column = lane16, lanes 0-15 K 0..7 & 16..23; lanes 16-31 K 8..15 & 24..31
        const lchar* p = Bb3 + (nW + ni * 16 + lane16) * ROWB + j * 64 + hseg * 16;
        bfr[ni].u[0] = *(const lv4u*)(p);
        bfr[ni].u[1] = *(const lv4u*)(p + 32);
      }
      #pragma unroll
      for (int mi = 0; mi < 2; ++mi)
        #pragma unroll
        for (int ni = 0; ni < 4; ++ni)
          acc[mi][ni] = __builtin_amdgcn_wmma_f32_16x16x32_bf16(
              false, af[mi].v, false, bfr[ni].v, (short)0, acc[mi][ni], false, false);
    }
  };

  const int nStages = K >> 6;
  issue_stage(0, 0);

  int s = 0;
  for (; s < nStages - 1; ++s) {               // steady state: branch-free wait
    issue_stage(s + 1, ((s + 1) & 1) * TILEB); // keep next slice in flight (16 ops out)
    WAIT_ASYNC(8);                             // oldest 8 (stage s) have landed
    __syncthreads();
    do_stage((s & 1) * TILEB);
    __syncthreads();
  }
  WAIT_ASYNC(0);                               // epilogue: last slice fully landed
  __syncthreads();
  do_stage((s & 1) * TILEB);

  const bool relu  = (flags & F_RELU)  != 0;
  const bool obf16 = (flags & F_OBF16) != 0;
  const bool trans = (flags & F_TRANS) != 0;

  // C/D 16x16 f32 layout: VGPR r -> (row = r + hseg*8, col = lane16)
  #pragma unroll
  for (int mi = 0; mi < 2; ++mi) {
    #pragma unroll
    for (int ni = 0; ni < 4; ++ni) {
      const int n = tileN0 + nW + ni * 16 + lane16;
      #pragma unroll
      for (int r = 0; r < 8; ++r) {
        const int m = tileM0 + mW + mi * 16 + hseg * 8 + r;
        float v = acc[mi][ni][r];
        if (relu && v < 0.0f) v = 0.0f;
        const long idx = (long)bz * sDb +
                         (trans ? ((long)n * ldd + m) : ((long)m * ldd + n));
        if (obf16) ((unsigned short*)Dv)[idx] = f32_to_bf16(v);
        else       ((float*)Dv)[idx]          = v;
      }
    }
  }
}

__global__ __launch_bounds__(256)
void cvt_f32_to_bf16(const float* __restrict__ in, unsigned short* __restrict__ out, long n) {
  const long i = ((long)blockIdx.x * 256 + threadIdx.x) * 4;
  if (i + 3 < n) {
    const float4 f = *(const float4*)(in + i);
    typedef __attribute__((ext_vector_type(4))) unsigned short v4us;
    v4us o;
    o.x = f32_to_bf16(f.x);
    o.y = f32_to_bf16(f.y);
    o.z = f32_to_bf16(f.z);
    o.w = f32_to_bf16(f.w);
    *(v4us*)(out + i) = o;
  }
}

extern "C" void kernel_launch(void* const* d_in, const int* in_sizes, int n_in,
                              void* d_out, int out_size, void* d_ws, size_t ws_size,
                              hipStream_t stream) {
  (void)in_sizes; (void)n_in; (void)out_size; (void)ws_size;
  const float* x  = (const float*)d_in[0];
  const float* Wa = (const float*)d_in[1];
  const float* Wb = (const float*)d_in[2];
  const float* Wc = (const float*)d_in[3];
  const float* Wd = (const float*)d_in[4];

  const long D = 1024, S = 4096, Bn = 4, BS = Bn * S;  // BS = 16384
  const long MB = 1024l * 1024l;

  char* ws = (char*)d_ws;
  // Layout (bytes):
  //   [0,128M)   : xb (32 MiB, dead after projections), then cmp (128 MiB bf16)
  //   [128,160M) : Abuf (32 MiB), reused as outbuf after cmp stage
  //   [160,192M) : Bbuf (32 MiB)
  //   [192,224M) : Ct   (32 MiB, C stored transposed per batch: [b][e][t])
  //   [224,232M) : bf16 weights wa..wd (2 MiB each)
  unsigned short* xb   = (unsigned short*)(ws);
  unsigned short* cmp  = (unsigned short*)(ws);
  unsigned short* Abuf = (unsigned short*)(ws + 128 * MB);
  unsigned short* outb = Abuf;
  unsigned short* Bbuf = (unsigned short*)(ws + 160 * MB);
  unsigned short* Ct   = (unsigned short*)(ws + 192 * MB);
  unsigned short* wa   = (unsigned short*)(ws + 224 * MB);
  unsigned short* wb   = wa + D * D;
  unsigned short* wc   = wb + D * D;
  unsigned short* wd   = wc + D * D;

  const dim3 blk(256);

  // f32 -> bf16 conversions
  cvt_f32_to_bf16<<<dim3((unsigned)((BS * D) / 1024)), blk, 0, stream>>>(x,  xb, BS * D);
  cvt_f32_to_bf16<<<dim3((unsigned)((D * D) / 1024)),  blk, 0, stream>>>(Wa, wa, D * D);
  cvt_f32_to_bf16<<<dim3((unsigned)((D * D) / 1024)),  blk, 0, stream>>>(Wb, wb, D * D);
  cvt_f32_to_bf16<<<dim3((unsigned)((D * D) / 1024)),  blk, 0, stream>>>(Wc, wc, D * D);
  cvt_f32_to_bf16<<<dim3((unsigned)((D * D) / 1024)),  blk, 0, stream>>>(Wd, wd, D * D);

  // A = relu(x Wa^T)   [16384 x 1024]
  gemm_nt<<<dim3(8, 128, 1), blk, 0, stream>>>(xb, wa, Abuf, 1024, 0, 0, 0, 1024,
                                               F_RELU | F_OBF16);
  // Bm = relu(x Wb^T)  [16384 x 1024]
  gemm_nt<<<dim3(8, 128, 1), blk, 0, stream>>>(xb, wb, Bbuf, 1024, 0, 0, 0, 1024,
                                               F_RELU | F_OBF16);
  // Ct[b] = relu(x_b Wc^T)^T  stored [1024 x 4096] per batch
  gemm_nt<<<dim3(8, 32, 4), blk, 0, stream>>>(xb, wc, Ct, 1024, S * D, 0, D * S, 4096,
                                              F_RELU | F_OBF16 | F_TRANS);
  // cmp[b] = A_b Bm_b^T   [4096 x 4096], K = 1024
  gemm_nt<<<dim3(32, 32, 4), blk, 0, stream>>>(Abuf, Bbuf, cmp, 1024, S * D, S * D, S * S,
                                               4096, F_OBF16);
  // out[b] = cmp_b Ct_b^T  [4096 x 1024], K = 4096
  gemm_nt<<<dim3(8, 32, 4), blk, 0, stream>>>(cmp, Ct, outb, 4096, S * S, D * S, S * D,
                                              1024, F_OBF16);
  // result = relu(out Wd^T)  [16384 x 1024] f32 -> d_out
  gemm_nt<<<dim3(8, 128, 1), blk, 0, stream>>>(outb, wd, d_out, 1024, 0, 0, 0, 1024,
                                               F_RELU);
}
